// Block_13829794693728
// MI455X (gfx1250) — compile-verified, matrix operand order checked by reading, not run on hardware
//
#include <hip/hip_runtime.h>
#include <math.h>

// ---------------------------------------------------------------------------
// Transformer block (pre-LN attention + pre-LN GELU MLP), B=1024 T=96 C=512
// H=4 D=128 F=2048, fp32 in/out. Memory-bound (~640 GFLOP vs ~2GB traffic at
// 23.3 TB/s) -> bf16 WMMA matmuls w/ fp32 accumulation, heavy fusion,
// double-buffered GEMM tile pipeline.
// ---------------------------------------------------------------------------

#define Bb 1024
#define Tt 96
#define Cc 512
#define Hh 4
#define Dd 128
#define Ff 2048
#define NTOK (Bb * Tt)          // 98304
#define MCHUNK 12288            // MLP row chunk (8 chunks)

typedef __attribute__((ext_vector_type(8)))  __bf16 v8bf;
typedef __attribute__((ext_vector_type(16))) __bf16 v16bf;
typedef __attribute__((ext_vector_type(8)))  float  v8f;

__device__ __forceinline__ __bf16 f2bf(float f) {
    unsigned u = __builtin_bit_cast(unsigned, f);
    unsigned r = u + 0x7FFFu + ((u >> 16) & 1u);   // round-to-nearest-even
    unsigned short h = (unsigned short)(r >> 16);
    return __builtin_bit_cast(__bf16, h);
}

__device__ __forceinline__ v8f vzero() {
    v8f z;
    #pragma unroll
    for (int i = 0; i < 8; ++i) z[i] = 0.0f;
    return z;
}

// Load one 16x32 bf16 fragment (A layout, or B^T layout from [N][K] storage).
// Per ISA 7.12.2: lane L holds row (L&15); lanes<16 hold K {0..7,16..23},
// lanes>=16 hold K {8..15,24..31}. Two 16B loads per lane.
__device__ __forceinline__ v16bf load_frag(const __bf16* base, int ld) {
    const int lane = threadIdx.x & 31;
    const __bf16* p = base + (size_t)(lane & 15) * ld + ((lane >> 4) << 3);
    v8bf lo = *(const v8bf*)p;
    v8bf hi = *(const v8bf*)(p + 16);
    return __builtin_shufflevector(lo, hi, 0,1,2,3,4,5,6,7,8,9,10,11,12,13,14,15);
}

__device__ __forceinline__ v8f wmma_bf16(v16bf a, v16bf b, v8f c) {
    return __builtin_amdgcn_wmma_f32_16x16x32_bf16(
        /*neg_a=*/false, a, /*neg_b=*/false, b,
        /*c_mod=*/(short)0, c, /*reuse_a=*/false, /*reuse_b=*/false);
}

// ---------------------------------------------------------------------------
// Per-row LayerNorm statistics: stats[row] = (mu, 1/sqrt(var+eps)), 512 cols.
// 8 rows per 256-thread block, one wave32 per row.
// ---------------------------------------------------------------------------
__global__ __launch_bounds__(256) void ln_stats_kernel(
    const float* __restrict__ x, float2* __restrict__ st)
{
    const int lane = threadIdx.x & 31;
    const size_t row = (size_t)blockIdx.x * 8 + (threadIdx.x >> 5);
    const float* p = x + row * Cc;
    float s = 0.f, ss = 0.f;
    #pragma unroll
    for (int j = 0; j < 16; ++j) {
        float v = p[lane + 32 * j];
        s += v; ss += v * v;
    }
    #pragma unroll
    for (int o = 16; o; o >>= 1) {
        s  += __shfl_xor(s,  o, 32);
        ss += __shfl_xor(ss, o, 32);
    }
    if (lane == 0) {
        float mu  = s * (1.0f / Cc);
        float var = ss * (1.0f / Cc) - mu * mu;
        st[row] = make_float2(mu, rsqrtf(var + 1e-5f));
    }
}

// ---------------------------------------------------------------------------
// fp32 [batch][R][S] -> bf16 transposed [batch][S][R]  (weight prep, ~6MB)
// ---------------------------------------------------------------------------
__global__ __launch_bounds__(256) void transpose_bf16_kernel(
    const float* __restrict__ in, __bf16* __restrict__ out,
    int batch, int R, int S)
{
    size_t total = (size_t)batch * R * S;
    for (size_t i = (size_t)blockIdx.x * blockDim.x + threadIdx.x; i < total;
         i += (size_t)gridDim.x * blockDim.x) {
        size_t bi  = i / ((size_t)R * S);
        size_t rem = i - bi * (size_t)R * S;
        int r = (int)(rem / S), s = (int)(rem % S);
        out[bi * (size_t)R * S + (size_t)s * R + r] = f2bf(in[i]);
    }
}

// ---------------------------------------------------------------------------
// Fully fused causal attention: one block per (b,h). LN1 -> xn (LDS, bf16)
// -> Q,K,V via WMMA -> causal QK^T (masked tiles skipped) -> softmax ->
// P@V -> o_ws[b*96+t][h*128+d] (bf16). q/k/v/scores never touch HBM.
// LDS: 96KB xn (aliased by scores+P) + 3*24KB = 168KB of the WGP's 320KB.
// ---------------------------------------------------------------------------
struct AttnSmem {
    union U {
        __bf16 xn[96][512];                           // 96KB  (phase 1-2)
        struct P2 {
            float  sc[96][100];                       // 38.4KB (phase 3)
            __bf16 P[96][96];                         // 18KB   (phase 4-5)
        } p2;
    } u;
    __bf16 Q [96][128];                               // A layout  [t][d]
    __bf16 K [96][128];                               // B^T layout [s][d]
    __bf16 VT[128][96];                               // B^T layout [d][s]
};

__global__ __launch_bounds__(256) void attn_kernel(
    const float* __restrict__ x, const float2* __restrict__ st1,
    const float* __restrict__ g1, const float* __restrict__ be1,
    const __bf16* __restrict__ wqkvT,   // [3][H][128][512] bf16 (D-major)
    __bf16* __restrict__ o_ws)          // [NTOK][512] bf16
{
    __shared__ AttnSmem sm;
    const int b = blockIdx.x >> 2, h = blockIdx.x & 3;
    const int tid = threadIdx.x, wave = tid >> 5, lane = tid & 31;
    const float SCALE = 0.08838834764831845f;   // D^-0.5

    // ---- phase 1: stage xn = LN1(x[b]) as bf16 into LDS -------------------
    const float* xb = x + (size_t)b * Tt * Cc;
    for (int i = tid; i < Tt * Cc; i += 256) {
        int r = i >> 9, c = i & 511;
        float2 s = st1[(size_t)b * Tt + r];
        sm.u.xn[r][c] = f2bf((xb[i] - s.x) * s.y * g1[c] + be1[c]);
    }
    __syncthreads();

    // ---- phase 2: Q/K/V = xn @ W  (M=96, N=128, K=512); wave -> N-tile ----
    for (int mat = 0; mat < 3; ++mat) {
        const __bf16* WT = wqkvT + ((size_t)(mat * Hh + h)) * Dd * Cc;
        v8f acc[6];
        #pragma unroll
        for (int mt = 0; mt < 6; ++mt) acc[mt] = vzero();
        for (int k0 = 0; k0 < Cc; k0 += 32) {
            v16bf bf = load_frag(WT + (size_t)(16 * wave) * Cc + k0, Cc);
            #pragma unroll
            for (int mt = 0; mt < 6; ++mt) {
                v16bf af = load_frag(&sm.u.xn[16 * mt][k0], 512);
                acc[mt] = wmma_bf16(af, bf, acc[mt]);
            }
        }
        #pragma unroll
        for (int mt = 0; mt < 6; ++mt)
            #pragma unroll
            for (int r = 0; r < 8; ++r) {
                int row = 16 * mt + r + 8 * (lane >> 4);
                int col = 16 * wave + (lane & 15);
                __bf16 hv = f2bf(acc[mt][r]);
                if      (mat == 0) sm.Q[row][col]  = hv;
                else if (mat == 1) sm.K[row][col]  = hv;
                else               sm.VT[col][row] = hv;   // store V transposed
            }
    }
    __syncthreads();   // xn dead; Q/K/VT ready; union now hosts sc/P

    // ---- phase 3: scores = (Q K^T)*SCALE, only tiles touching causal region
    {
        int cnt = 0;
        for (int tt = 0; tt < 6; ++tt)
            for (int stt = 0; stt <= tt; ++stt, ++cnt) {
                if ((cnt & 7) != wave) continue;          // wave-uniform
                v8f a = vzero();
                #pragma unroll
                for (int k0 = 0; k0 < Dd; k0 += 32) {
                    v16bf qf = load_frag(&sm.Q[16 * tt][k0], 128);
                    v16bf kf = load_frag(&sm.K[16 * stt][k0], 128);
                    a = wmma_bf16(qf, kf, a);
                }
                #pragma unroll
                for (int r = 0; r < 8; ++r) {
                    int row = 16 * tt + r + 8 * (lane >> 4);
                    int col = 16 * stt + (lane & 15);
                    sm.u.p2.sc[row][col] = a[r] * SCALE;
                }
            }
    }
    __syncthreads();

    // ---- phase 4: causal softmax per row; P bf16 (masked cols -> 0) -------
    for (int i = 0; i < 12; ++i) {
        int row = wave + 8 * i;
        float v[3], e[3], mx = -3.0e38f;
        #pragma unroll
        for (int j = 0; j < 3; ++j) {
            int col = lane + 32 * j;
            v[j] = (col <= row) ? sm.u.p2.sc[row][col] : -3.0e38f;
            mx = fmaxf(mx, v[j]);
        }
        #pragma unroll
        for (int o = 16; o; o >>= 1) mx = fmaxf(mx, __shfl_xor(mx, o, 32));
        float s = 0.f;
        #pragma unroll
        for (int j = 0; j < 3; ++j) {
            int col = lane + 32 * j;
            e[j] = (col <= row) ? __expf(v[j] - mx) : 0.0f;
            s += e[j];
        }
        #pragma unroll
        for (int o = 16; o; o >>= 1) s += __shfl_xor(s, o, 32);
        float inv = 1.0f / s;
        #pragma unroll
        for (int j = 0; j < 3; ++j)
            sm.u.p2.P[row][lane + 32 * j] = f2bf(e[j] * inv);
    }
    __syncthreads();

    // ---- phase 5: O = P @ V  (M=96, N=128, K=96); wave -> N-tile ----------
    {
        v8f acc[6];
        #pragma unroll
        for (int mt = 0; mt < 6; ++mt) acc[mt] = vzero();
        #pragma unroll
        for (int k0 = 0; k0 < 96; k0 += 32) {
            v16bf bf = load_frag(&sm.VT[16 * wave][k0], 96);
            #pragma unroll
            for (int mt = 0; mt < 6; ++mt) {
                v16bf af = load_frag(&sm.u.p2.P[16 * mt][k0], 96);
                acc[mt] = wmma_bf16(af, bf, acc[mt]);
            }
        }
        #pragma unroll
        for (int mt = 0; mt < 6; ++mt)
            #pragma unroll
            for (int r = 0; r < 8; ++r) {
                int trow = 16 * mt + r + 8 * (lane >> 4);
                int col  = 16 * wave + (lane & 15);
                size_t tok = (size_t)b * Tt + trow;
                o_ws[tok * Cc + (size_t)h * Dd + col] = f2bf(acc[mt][r]);
            }
    }
}

// ---------------------------------------------------------------------------
// Generic 128x128-tile bf16 WMMA GEMM, K-step 32, DOUBLE-BUFFERED LDS:
// prefetch K-step k+1 from global into VGPRs while WMMAs consume buffer p,
// ds_store into buffer p^1, single __syncthreads per K-step.
// 8 waves, each owns a 32x64 sub-tile = 2x4 accumulators of 16x16.
// AMODE 0: A is bf16 [M][K].  AMODE 1: A is fp32 [M][K] with LN applied
//          during staging (lnst row stats + per-col gamma/beta).
// EPI   0: out fp32 = acc + bias + residual.
// EPI   1: out bf16 = GELU_exact(acc + bias).
// ---------------------------------------------------------------------------
template<int AMODE, int EPI>
__global__ __launch_bounds__(256) void gemm_tile_kernel(
    const void* __restrict__ Asrc, const __bf16* __restrict__ BT,
    const float* __restrict__ bias, const float* __restrict__ residual,
    void* __restrict__ outp, int N, int K,
    const float2* __restrict__ lnst, const float* __restrict__ lg,
    const float* __restrict__ lb)
{
    __shared__ __bf16 At[2][128][32];
    __shared__ __bf16 Bt[2][128][32];   // B^T layout: [n][k]
    const int tid = threadIdx.x, wave = tid >> 5, lane = tid & 31;
    const int wm = wave & 3, wn = wave >> 2;
    const size_t m0 = (size_t)blockIdx.x * 128;
    const int n0 = blockIdx.y * 128;

    // Per-thread staging: chunks (rr, part) and (rr+64, part), 8 bf16 each.
    const int rr = tid >> 2, part = (tid & 3) * 8;

    // LN row stats are loop-invariant for AMODE 1.
    float2 st0 = make_float2(0.f, 0.f), st1 = st0;
    if (AMODE == 1) { st0 = lnst[m0 + rr]; st1 = lnst[m0 + rr + 64]; }

    v8bf rA0, rA1, rB0, rB1;
    auto fetch = [&](int k0) {
        rB0 = *(const v8bf*)(BT + (size_t)(n0 + rr)      * K + k0 + part);
        rB1 = *(const v8bf*)(BT + (size_t)(n0 + rr + 64) * K + k0 + part);
        if (AMODE == 0) {
            const __bf16* ab = (const __bf16*)Asrc;
            rA0 = *(const v8bf*)(ab + (m0 + rr)      * (size_t)K + k0 + part);
            rA1 = *(const v8bf*)(ab + (m0 + rr + 64) * (size_t)K + k0 + part);
        } else {
            const float* af = (const float*)Asrc;
            const float* a0 = af + (m0 + rr)      * (size_t)K + k0 + part;
            const float* a1 = af + (m0 + rr + 64) * (size_t)K + k0 + part;
            #pragma unroll
            for (int j = 0; j < 8; ++j) {
                int col = k0 + part + j;
                rA0[j] = f2bf((a0[j] - st0.x) * st0.y * lg[col] + lb[col]);
                rA1[j] = f2bf((a1[j] - st1.x) * st1.y * lg[col] + lb[col]);
            }
        }
    };
    auto stage = [&](int p) {
        *(v8bf*)&At[p][rr][part]      = rA0;
        *(v8bf*)&At[p][rr + 64][part] = rA1;
        *(v8bf*)&Bt[p][rr][part]      = rB0;
        *(v8bf*)&Bt[p][rr + 64][part] = rB1;
    };

    v8f acc[2][4];
    #pragma unroll
    for (int mi = 0; mi < 2; ++mi)
        #pragma unroll
        for (int ni = 0; ni < 4; ++ni) acc[mi][ni] = vzero();

    fetch(0);
    stage(0);
    __syncthreads();

    int p = 0;
    for (int k0 = 0; k0 < K; k0 += 32) {
        const bool has_next = (k0 + 32 < K);
        if (has_next) fetch(k0 + 32);               // overlaps with WMMAs below
        v16bf bf[4];
        #pragma unroll
        for (int ni = 0; ni < 4; ++ni)
            bf[ni] = load_frag(&Bt[p][64 * wn + 16 * ni][0], 32);
        #pragma unroll
        for (int mi = 0; mi < 2; ++mi) {
            v16bf af = load_frag(&At[p][32 * wm + 16 * mi][0], 32);
            #pragma unroll
            for (int ni = 0; ni < 4; ++ni)
                acc[mi][ni] = wmma_bf16(af, bf[ni], acc[mi][ni]);
        }
        if (has_next) stage(p ^ 1);
        __syncthreads();
        p ^= 1;
    }

    #pragma unroll
    for (int mi = 0; mi < 2; ++mi)
        #pragma unroll
        for (int ni = 0; ni < 4; ++ni)
            #pragma unroll
            for (int r = 0; r < 8; ++r) {
                size_t m = m0 + 32 * wm + 16 * mi + r + 8 * (lane >> 4);
                int    n = n0 + 64 * wn + 16 * ni + (lane & 15);
                float  v = acc[mi][ni][r] + bias[n];
                size_t idx = m * (size_t)N + n;
                if (EPI == 0) {
                    ((float*)outp)[idx] = v + residual[idx];
                } else {
                    float gl = 0.5f * v * (1.0f + erff(v * 0.70710678118654752f));
                    ((__bf16*)outp)[idx] = f2bf(gl);
                }
            }
}

// ---------------------------------------------------------------------------
// Host-side orchestration. Workspace layout (~158 MB):
//   stats1 | stats2 | wqkvT | wprojT | w1T | w2T | o_ws | h_ws
// d_out doubles as the x1 residual buffer (proj writes it, fc2 updates it
// in place), so no extra 192MB fp32 buffer is needed.
// ---------------------------------------------------------------------------
extern "C" void kernel_launch(void* const* d_in, const int* in_sizes, int n_in,
                              void* d_out, int out_size, void* d_ws, size_t ws_size,
                              hipStream_t stream) {
    (void)in_sizes; (void)n_in; (void)out_size; (void)ws_size;
    const float* x      = (const float*)d_in[0];
    const float* wq     = (const float*)d_in[1];
    const float* wk     = (const float*)d_in[2];
    const float* wv     = (const float*)d_in[3];
    const float* w_proj = (const float*)d_in[4];
    const float* b_proj = (const float*)d_in[5];
    const float* w1     = (const float*)d_in[6];
    const float* b1     = (const float*)d_in[7];
    const float* w2     = (const float*)d_in[8];
    const float* b2     = (const float*)d_in[9];
    const float* g1     = (const float*)d_in[10];
    const float* be1    = (const float*)d_in[11];
    const float* g2     = (const float*)d_in[12];
    const float* be2    = (const float*)d_in[13];

    char* ws = (char*)d_ws;
    float2* stats1 = (float2*)ws;               ws += (size_t)NTOK * 8;
    float2* stats2 = (float2*)ws;               ws += (size_t)NTOK * 8;
    __bf16* wqkvT  = (__bf16*)ws;               ws += (size_t)3 * Hh * Dd * Cc * 2;
    __bf16* wprojT = (__bf16*)ws;               ws += (size_t)Cc * Cc * 2;
    __bf16* w1T    = (__bf16*)ws;               ws += (size_t)Ff * Cc * 2;
    __bf16* w2T    = (__bf16*)ws;               ws += (size_t)Cc * Ff * 2;
    __bf16* o_ws   = (__bf16*)ws;               ws += (size_t)NTOK * Cc * 2;
    __bf16* h_ws   = (__bf16*)ws;               // MCHUNK * F * 2 = 50MB

    // Weight prep (bf16, [N][K] layout). Tiny; grid-stride.
    transpose_bf16_kernel<<<1024, 256, 0, stream>>>(wq, wqkvT,                Hh, Cc, Dd);
    transpose_bf16_kernel<<<1024, 256, 0, stream>>>(wk, wqkvT + (size_t)Hh*Dd*Cc,   Hh, Cc, Dd);
    transpose_bf16_kernel<<<1024, 256, 0, stream>>>(wv, wqkvT + (size_t)2*Hh*Dd*Cc, Hh, Cc, Dd);
    transpose_bf16_kernel<<<1024, 256, 0, stream>>>(w_proj, wprojT, 1, Cc, Cc);
    transpose_bf16_kernel<<<1024, 256, 0, stream>>>(w1, w1T, 1, Cc, Ff);
    transpose_bf16_kernel<<<1024, 256, 0, stream>>>(w2, w2T, 1, Ff, Cc);

    // LN1 stats, fused attention.
    ln_stats_kernel<<<NTOK / 8, 256, 0, stream>>>(x, stats1);
    attn_kernel<<<Bb * Hh, 256, 0, stream>>>(x, stats1, g1, be1, wqkvT, o_ws);

    // x1 = x + o @ w_proj + b_proj   (written into d_out)
    gemm_tile_kernel<0, 0><<<dim3(NTOK / 128, Cc / 128), 256, 0, stream>>>(
        o_ws, wprojT, b_proj, x, d_out, Cc, Cc, nullptr, nullptr, nullptr);

    // LN2 stats over x1.
    ln_stats_kernel<<<NTOK / 8, 256, 0, stream>>>((const float*)d_out, stats2);

    // MLP in 8 row-chunks (keeps hidden buffer at 50MB):
    //   h = GELU(LN2(x1) @ w1 + b1);  x_out = x1 + h @ w2 + b2 (in place)
    for (int c = 0; c < NTOK / MCHUNK; ++c) {
        const size_t roff = (size_t)c * MCHUNK;
        const float* x1c = (const float*)d_out + roff * Cc;
        gemm_tile_kernel<1, 1><<<dim3(MCHUNK / 128, Ff / 128), 256, 0, stream>>>(
            x1c, w1T, b1, nullptr, h_ws, Ff, Cc, stats2 + roff, g2, be2);
        gemm_tile_kernel<0, 0><<<dim3(MCHUNK / 128, Cc / 128), 256, 0, stream>>>(
            h_ws, w2T, b2, x1c, (float*)d_out + roff * Cc, Cc, Ff,
            nullptr, nullptr, nullptr);
    }
}
